// Encoder_28595892256997
// MI455X (gfx1250) — matrix operand order repaired and checked by reading, not compiled, not run on hardware
//
#include <hip/hip_runtime.h>
#include <hip/hip_bf16.h>
#include <stdint.h>

#define BATCH 8
#define NN3 131072
#define NN2 65536
#define NN1 1024

typedef __attribute__((ext_vector_type(16))) __bf16 v16bf;
typedef __attribute__((ext_vector_type(8)))  float  v8f;

union FragB { v16bf v; uint4 q[2]; };

// ---------------------------------------------------------------------------
// Prep: Wb [128(n)][256(k)][3(v)] f32  ->  Wt [3(v)][128(n)][256(k)] bf16
// ---------------------------------------------------------------------------
__global__ void prep_bilinear_w(const float* __restrict__ W,
                                __hip_bfloat16* __restrict__ Wt) {
    int f = blockIdx.x * 256 + threadIdx.x;           // 3*128*256 = 98304
    if (f >= 3 * 128 * 256) return;
    int v   = f / (128 * 256);
    int rem = f - v * 128 * 256;
    int n   = rem >> 8;
    int k   = rem & 255;
    Wt[f] = __float2bfloat16(W[(size_t)n * 768 + k * 3 + v]);
}

__global__ void prep_linear_w(const float* __restrict__ W,
                              __hip_bfloat16* __restrict__ Wt, int n) {
    int i = blockIdx.x * 256 + threadIdx.x;
    if (i < n) Wt[i] = __float2bfloat16(W[i]);
}

// ---------------------------------------------------------------------------
// Leaf layer: h3 = prelu(points @ Wp^T + bp)   (K=3 -> VALU kernel)
// ---------------------------------------------------------------------------
__global__ __launch_bounds__(256)
void leaf_kernel(const float* __restrict__ pts, const float* __restrict__ Wp,
                 const float* __restrict__ bp, const float* __restrict__ ap,
                 __hip_bfloat16* __restrict__ h3, int totalNodes) {
    __shared__ float Wsh[384];
    __shared__ float bsh[128];
    int tid = threadIdx.x;
    for (int i = tid; i < 384; i += 256) Wsh[i] = Wp[i];
    for (int i = tid; i < 128; i += 256) bsh[i] = bp[i];
    __syncthreads();

    int gid  = blockIdx.x * 256 + tid;
    int node = gid >> 2;
    if (node >= totalNodes) return;
    int k0 = (gid & 3) * 32;

    const float* p = pts + (size_t)node * 3;
    float p0 = p[0], p1 = p[1], p2 = p[2];
    float a = *ap;

    union { uint4 q[4]; unsigned u[16]; } outv;
#pragma unroll
    for (int i = 0; i < 16; ++i) {
        unsigned w = 0;
#pragma unroll
        for (int h = 0; h < 2; ++h) {
            int k = k0 + 2 * i + h;
            float z = p0 * Wsh[k * 3] + p1 * Wsh[k * 3 + 1] + p2 * Wsh[k * 3 + 2] + bsh[k];
            z = (z >= 0.f) ? z : a * z;
            __hip_bfloat16 hb = __float2bfloat16(z);
            unsigned short us = *reinterpret_cast<unsigned short*>(&hb);
            w |= ((unsigned)us) << (16 * h);
        }
        outv.u[i] = w;
    }
    uint4* dst = (uint4*)((unsigned short*)h3 + (size_t)node * 128 + k0);
#pragma unroll
    for (int i = 0; i < 4; ++i) dst[i] = outv.q[i];
}

// ---------------------------------------------------------------------------
// Gather + GEMM (WMMA bf16, f32 accum).
//   NV==3: bilinear  out = prelu( sum_v (A @ Wt_v) * vec[.,v] + bias )
//   NV==1: linear    out = prelu( A @ Wt + bias )
// A rows: concat( srcL[idxL[n] or n], srcR[idxR[n]] )  (128+128 bf16)
// Wave w owns N-tile w; caches its NV*8 B-fragments in registers across a
// loop of M-tiles (B-matrix L2 traffic amortized).
// ---------------------------------------------------------------------------
template<int NV, bool IDENT_LEFT>
__global__ __launch_bounds__(256)
void merge_gemm(const __hip_bfloat16* __restrict__ srcL,
                const __hip_bfloat16* __restrict__ srcR,
                const int* __restrict__ idxL,
                const int* __restrict__ idxR,
                const float* __restrict__ vec,
                const __hip_bfloat16* __restrict__ Wt,   // [NV][128][256] bf16
                const float* __restrict__ bias,
                const float* __restrict__ aptr,
                __hip_bfloat16* __restrict__ hout,
                int nnodeShift, int nsrcL, int nsrcR, int tilesPerBlock) {
    __shared__ __align__(16) unsigned short As[16 * 264];  // padded rows (bank spread)
    __shared__ float Vs[48];

    const int tid   = threadIdx.x;
    const int wave  = tid >> 5;
    const int lane  = tid & 31;
    const int g     = lane >> 4;
    const int col16 = lane & 15;
    const int col   = wave * 16 + col16;
    const int nnodeMask = (1 << nnodeShift) - 1;

    // Cache B fragments in registers. B layout used here (32x16 bf16): lane holds
    // column N=lane%16, K = 16*(lane/16) .. +16 -> 16 contiguous K values.
    FragB Bf[NV][8];
#pragma unroll
    for (int v = 0; v < NV; ++v)
#pragma unroll
        for (int s = 0; s < 8; ++s) {
            const uint4* p = (const uint4*)(Wt + ((size_t)(v * 128 + col) * 256 + s * 32 + 16 * g));
            Bf[v][s].q[0] = p[0];
            Bf[v][s].q[1] = p[1];
        }

    const float bcol = bias[col];
    const float aval = *aptr;

    const int r16 = tid >> 4;   // gather row 0..15
    const int c16 = tid & 15;   // gather 16-elem chunk 0..15

    for (int t = 0; t < tilesPerBlock; ++t) {
        const int m_base = (blockIdx.x * tilesPerBlock + t) * 16;

        { // gather A tile: 16 rows x 256 bf16 (each thread moves 32 bytes)
            const int m = m_base + r16;
            const int b = m >> nnodeShift;
            const int n = m & nnodeMask;
            const uint4* sp;
            if (c16 < 8) {
                int s0;
                if constexpr (IDENT_LEFT) s0 = n; else s0 = idxL[n];
                sp = (const uint4*)((const unsigned short*)srcL +
                                    ((size_t)b * nsrcL + s0) * 128 + c16 * 16);
            } else {
                int s1 = idxR[n];
                sp = (const uint4*)((const unsigned short*)srcR +
                                    ((size_t)b * nsrcR + s1) * 128 + (c16 - 8) * 16);
            }
            uint4 x0 = sp[0];
            uint4 x1 = sp[1];
            uint4* dp = (uint4*)(&As[r16 * 264 + c16 * 16]);
            dp[0] = x0; dp[1] = x1;
        }
        if constexpr (NV == 3) {
            if (tid < 48) Vs[tid] = vec[(size_t)m_base * 3 + tid];
        }
        __syncthreads();

        v8f acc[NV];
#pragma unroll
        for (int v = 0; v < NV; ++v)
#pragma unroll
            for (int i = 0; i < 8; ++i) acc[v][i] = 0.f;

#pragma unroll
        for (int s = 0; s < 8; ++s) {
            // A layout (16x32 bf16): lane row M=lane%16; regs 0-7 hold
            // K = 16g..16g+15 (contiguous, matching B)  -> two b128 reads.
            FragB Af;
            const uint4* apv = (const uint4*)(&As[col16 * 264 + s * 32 + 16 * g]);
            Af.q[0] = apv[0];
            Af.q[1] = apv[1];
#pragma unroll
            for (int v = 0; v < NV; ++v)
                acc[v] = __builtin_amdgcn_wmma_f32_16x16x32_bf16(
                    false, Af.v, false, Bf[v][s].v, (short)0, acc[v], false, false);
        }

        // epilogue: scale chains by per-row vec, bias, PReLU, store bf16
#pragma unroll
        for (int r = 0; r < 8; ++r) {
            const int Mrow = r + 8 * g;
            float z;
            if constexpr (NV == 3) {
                z = acc[0][r] * Vs[Mrow * 3 + 0]
                  + acc[1][r] * Vs[Mrow * 3 + 1]
                  + acc[2][r] * Vs[Mrow * 3 + 2] + bcol;
            } else {
                z = acc[0][r] + bcol;
            }
            z = (z >= 0.f) ? z : aval * z;
            hout[(size_t)(m_base + Mrow) * 128 + col] = __float2bfloat16(z);
        }
        __syncthreads();
    }
}

// ---------------------------------------------------------------------------
// Root layer (M = 8 rows): plain fp32 VALU, one block.
// ---------------------------------------------------------------------------
__global__ __launch_bounds__(256)
void root_kernel(const __hip_bfloat16* __restrict__ h1,
                 const __hip_bfloat16* __restrict__ h2,
                 const float* __restrict__ vec0, const float* __restrict__ Wb0,
                 const float* __restrict__ bb0, const float* __restrict__ Ws0,
                 const float* __restrict__ bs0, const float* __restrict__ a0p,
                 const int* __restrict__ cl0, const int* __restrict__ cr0,
                 const int* __restrict__ cs0, float* __restrict__ out) {
    __shared__ float x0[256];
    __shared__ float hc[256];
    int tid = threadIdx.x;
    float a0 = *a0p;
    int il = cl0[0], ir = cr0[0], is = cs0[0];
    for (int b = 0; b < 8; ++b) {
        {
            int d = tid;
            int src = (d < 128) ? il : ir;
            x0[d] = __bfloat162float(h1[((size_t)b * NN1 + src) * 128 + (d & 127)]);
        }
        __syncthreads();
        if (tid < 128) {
            int k = tid;
            float acc = bb0[k];
            for (int v = 0; v < 3; ++v) {
                float sv = 0.f;
                for (int d = 0; d < 256; ++d)
                    sv += x0[d] * Wb0[(size_t)k * 768 + d * 3 + v];
                acc += sv * vec0[b * 3 + v];
            }
            acc = (acc >= 0.f) ? acc : a0 * acc;
            hc[k] = acc;
        } else {
            int d = tid - 128;
            hc[128 + d] = __bfloat162float(h2[((size_t)b * NN2 + is) * 128 + d]);
        }
        __syncthreads();
        if (tid < 128) {
            int k = tid;
            float acc = bs0[k];
            for (int d = 0; d < 256; ++d)
                acc += hc[d] * Ws0[(size_t)k * 256 + d];
            acc = (acc >= 0.f) ? acc : a0 * acc;
            out[b * 128 + k] = acc;
        }
        __syncthreads();
    }
}

// ---------------------------------------------------------------------------
extern "C" void kernel_launch(void* const* d_in, const int* in_sizes, int n_in,
                              void* d_out, int out_size, void* d_ws, size_t ws_size,
                              hipStream_t stream) {
    const float* points = (const float*)d_in[0];
    const float* vec2   = (const float*)d_in[1];
    const float* vec1   = (const float*)d_in[2];
    const float* vec0   = (const float*)d_in[3];
    const float* Wp     = (const float*)d_in[4];
    const float* bp     = (const float*)d_in[5];
    const float* a_leaf = (const float*)d_in[6];
    const float* Wb2    = (const float*)d_in[7];
    const float* bb2    = (const float*)d_in[8];
    const float* a2     = (const float*)d_in[9];
    const float* Wb1    = (const float*)d_in[10];
    const float* bb1    = (const float*)d_in[11];
    const float* Ws1    = (const float*)d_in[12];
    const float* bs1    = (const float*)d_in[13];
    const float* a1     = (const float*)d_in[14];
    const float* Wb0    = (const float*)d_in[15];
    const float* bb0    = (const float*)d_in[16];
    const float* Ws0    = (const float*)d_in[17];
    const float* bs0    = (const float*)d_in[18];
    const float* a0     = (const float*)d_in[19];
    const int* cl2 = (const int*)d_in[20];
    const int* cr2 = (const int*)d_in[21];
    const int* cl1 = (const int*)d_in[22];
    const int* cr1 = (const int*)d_in[23];
    const int* cs1 = (const int*)d_in[24];
    const int* cl0 = (const int*)d_in[25];
    const int* cr0 = (const int*)d_in[26];
    const int* cs0 = (const int*)d_in[27];

    char* ws = (char*)d_ws;
    size_t o_h3  = 0;
    size_t o_h2  = o_h3  + (size_t)BATCH * NN3 * 128 * 2;   // 256 MB
    size_t o_h1a = o_h2  + (size_t)BATCH * NN2 * 128 * 2;   // 128 MB
    size_t o_h1  = o_h1a + (size_t)BATCH * NN1 * 128 * 2;   // 2 MB
    size_t o_wt2 = o_h1  + (size_t)BATCH * NN1 * 128 * 2;   // 2 MB
    size_t o_wt1 = o_wt2 + (size_t)3 * 128 * 256 * 2;
    size_t o_ws1 = o_wt1 + (size_t)3 * 128 * 256 * 2;
    size_t need  = o_ws1 + (size_t)128 * 256 * 2;
    if (ws_size < need) return;

    __hip_bfloat16* h3    = (__hip_bfloat16*)(ws + o_h3);
    __hip_bfloat16* h2    = (__hip_bfloat16*)(ws + o_h2);
    __hip_bfloat16* h1a   = (__hip_bfloat16*)(ws + o_h1a);
    __hip_bfloat16* h1    = (__hip_bfloat16*)(ws + o_h1);
    __hip_bfloat16* wt2   = (__hip_bfloat16*)(ws + o_wt2);
    __hip_bfloat16* wt1   = (__hip_bfloat16*)(ws + o_wt1);
    __hip_bfloat16* ws1bf = (__hip_bfloat16*)(ws + o_ws1);

    // weight prep
    prep_bilinear_w<<<384, 256, 0, stream>>>(Wb2, wt2);
    prep_bilinear_w<<<384, 256, 0, stream>>>(Wb1, wt1);
    prep_linear_w<<<128, 256, 0, stream>>>(Ws1, ws1bf, 128 * 256);

    // leaf: B*N3 = 1,048,576 nodes, 4 threads/node
    leaf_kernel<<<(BATCH * NN3 * 4) / 256, 256, 0, stream>>>(
        points, Wp, bp, a_leaf, h3, BATCH * NN3);

    // layer 2: 32768 M-tiles, 16 per block -> 2048 blocks
    merge_gemm<3, false><<<2048, 256, 0, stream>>>(
        h3, h3, cl2, cr2, vec2, wt2, bb2, a2, h2, 16, NN3, NN3, 16);

    // layer 1 bilinear: 512 M-tiles, 2 per block -> 256 blocks
    merge_gemm<3, false><<<256, 256, 0, stream>>>(
        h2, h2, cl1, cr1, vec1, wt1, bb1, a1, h1a, 10, NN2, NN2, 2);

    // layer 1 sample-merge linear: concat(h1a[n], h3[cs1[n]]) @ Ws1^T
    merge_gemm<1, true><<<256, 256, 0, stream>>>(
        h1a, h3, nullptr, cs1, nullptr, ws1bf, bs1, a1, h1, 10, NN1, NN3, 2);

    // root
    root_kernel<<<1, 256, 0, stream>>>(h1, h2, vec0, Wb0, bb0, Ws0, bs0, a0,
                                       cl0, cr0, cs0, (float*)d_out);
}
